// GraphNetv2_46935402611088
// MI455X (gfx1250) — matrix operand order repaired and checked by reading, not compiled, not run on hardware
//
#include <hip/hip_runtime.h>
#include <stdint.h>

// MI455X / gfx1250, wave32. GraphNet fused pipeline:
//   prep (f32->f16 weight/x pre-layout, once) -> zero_stats ->
//   edge_mlp (WMMA f16, per (batch,receiver) WG, async LDS staging) ->
//   node_mlp (WMMA f16, async LDS staging).
// ws layout (floats): [0..19] sum(E), [20..39] sum(E^2), [64..) Ebar_raw (64,20,128),
// then f16 blob at float-offset 163904: W1e|W2e|W3e|W1o|W2o|W3o|xT.

typedef __attribute__((ext_vector_type(16))) _Float16 v16h;
typedef __attribute__((ext_vector_type(8)))  float    v8f;
typedef __attribute__((ext_vector_type(4)))  int      v4i;

#define NND   128
#define PFEAT 30
#define NB    64
#define EPN   127
#define EPS_F 1e-5f

// f16 blob offsets (in halves)
#define OFF_W1E 0        // 80x64  (edge L1, K split: [0,30)=recv, [32,62)=send)
#define OFF_W2E 5120     // 48x96  (edge L2)
#define OFF_W3E 9728     // 32x64  (edge L3)
#define OFF_W1O 11776    // 48x64  (node L1, K split: [0,30)=x, [32,52)=Ebar)
#define OFF_W2O 14848    // 32x64  (node L2)
#define OFF_W3O 16896    // 32x32  (node L3)
#define OFF_XT  17920    // 64 x 128 x 32 (xT[b][node][feat], feat padded to 32)
#define WS_HALF_OFF 163904   // float offset of f16 blob in ws

// ---- CDNA5 async global->LDS copy (16B per lane), ASYNCcnt tracked ---------
__device__ __forceinline__ void async_g2l_b128(_Float16* lds, const _Float16* g) {
#if __has_builtin(__builtin_amdgcn_global_load_async_to_lds_b128)
  typedef __attribute__((address_space(1))) v4i* gp_t;   // global src
  typedef __attribute__((address_space(3))) v4i* lp_t;   // LDS dst (32-bit offset)
  __builtin_amdgcn_global_load_async_to_lds_b128(
      (gp_t)(uintptr_t)g, (lp_t)(uintptr_t)lds, 0, 0);
#else
  const unsigned l = (unsigned)(uintptr_t)lds;           // low 32b = LDS offset
  const unsigned long long ga = (unsigned long long)(uintptr_t)g;
  asm volatile("global_load_async_to_lds_b128 %0, %1, off"
               :: "v"(l), "v"(ga) : "memory");
#endif
}
__device__ __forceinline__ void wait_async0() {
#if __has_builtin(__builtin_amdgcn_s_wait_asynccnt)
  __builtin_amdgcn_s_wait_asynccnt(0);
#else
  asm volatile("s_wait_asynccnt 0x0" ::: "memory");
#endif
}

// ---- WMMA fragment loaders per CDNA5 ISA 7.12.2 layouts --------------------
__device__ __forceinline__ v16h frag_a(const _Float16* w, int m0, int k0, int S, int lane) {
  v16h f;
  unsigned int* u = reinterpret_cast<unsigned int*>(&f);
  const int h = lane >> 4;
  const _Float16* p = w + (m0 + (lane & 15)) * S + k0;
#pragma unroll
  for (int j = 0; j < 8; ++j) {
    const int k = (j < 4) ? (2 * j + 8 * h) : (16 + 2 * (j - 4) + 8 * h);
    u[j] = *reinterpret_cast<const unsigned int*>(p + k);
  }
  return f;
}
__device__ __forceinline__ v16h frag_b(const _Float16* a, int c0, int k0, int S, int lane) {
  v16h f;
  unsigned int* u = reinterpret_cast<unsigned int*>(&f);
  const int h = lane >> 4;
  const _Float16* p = a + (c0 + (lane & 15)) * S + k0;
#pragma unroll
  for (int j = 0; j < 8; ++j) {
    const int k = (j < 4) ? (2 * j + 8 * h) : (16 + 2 * (j - 4) + 8 * h);
    u[j] = *reinterpret_cast<const unsigned int*>(p + k);
  }
  return f;
}

__global__ void zero_stats_kernel(float* p) {
  if (threadIdx.x < 40) p[threadIdx.x] = 0.f;
}

// ---- prep: weights -> padded f16 blobs; x -> f16 transpose xT[b][n][32] ----
__global__ __launch_bounds__(256) void prep_kernel(
    const float* __restrict__ x,
    const float* __restrict__ Wfr1, const float* __restrict__ Wfr2,
    const float* __restrict__ Wfr3,
    const float* __restrict__ Wfo1, const float* __restrict__ Wfo2,
    const float* __restrict__ Wfo3,
    _Float16* __restrict__ hb)
{
  const int tid = threadIdx.x, b = blockIdx.x;
  const float* xb = x + b * (PFEAT * NND);
  _Float16* xTb = hb + OFF_XT + b * (NND * 32);
  for (int i = tid; i < NND * 32; i += 256) {
    const int n = i >> 5, k = i & 31;
    xTb[i] = (_Float16)((k < 30) ? xb[k * NND + n] : 0.f);
  }
  if (b != 0) return;
  for (int i = tid; i < 80 * 64; i += 256) {          // W1e, K split at 32
    const int m = i >> 6, k = i & 63;
    float v = 0.f;
    if (k < 30)                 v = Wfr1[m * 60 + k];
    else if (k >= 32 && k < 62) v = Wfr1[m * 60 + (k - 2)];
    hb[OFF_W1E + i] = (_Float16)v;
  }
  for (int i = tid; i < 48 * 96; i += 256) {          // W2e
    const int m = i / 96, k = i - m * 96;
    hb[OFF_W2E + i] = (_Float16)((m < 40 && k < 80) ? Wfr2[m * 80 + k] : 0.f);
  }
  for (int i = tid; i < 32 * 64; i += 256) {          // W3e
    const int m = i >> 6, k = i & 63;
    hb[OFF_W3E + i] = (_Float16)((m < 20 && k < 40) ? Wfr3[m * 40 + k] : 0.f);
  }
  for (int i = tid; i < 48 * 64; i += 256) {          // W1o, K split at 32
    const int m = i >> 6, k = i & 63;
    float v = 0.f;
    if (m < 40) {
      if (k < 30)                 v = Wfo1[m * 50 + k];
      else if (k >= 32 && k < 52) v = Wfo1[m * 50 + (k - 2)];
    }
    hb[OFF_W1O + i] = (_Float16)v;
  }
  for (int i = tid; i < 32 * 64; i += 256) {          // W2o
    const int m = i >> 6, k = i & 63;
    hb[OFF_W2O + i] = (_Float16)((m < 20 && k < 40) ? Wfo2[m * 40 + k] : 0.f);
  }
  for (int i = tid; i < 32 * 32; i += 256) {          // W3o
    const int m = i >> 5, k = i & 31;
    hb[OFF_W3O + i] = (_Float16)((m < 20 && k < 20) ? Wfo3[m * 20 + k] : 0.f);
  }
}

// ---- Kernel A: edge MLP (60->80->40->20), stats, exclusive segment sums ----
__global__ __launch_bounds__(256) void edge_mlp_kernel(
    const _Float16* __restrict__ xT, const int* __restrict__ send,
    const _Float16* __restrict__ wblob,
    const float* __restrict__ bfr1, const float* __restrict__ bfr2,
    const float* __restrict__ bfr3,
    float* __restrict__ gsum, float* __restrict__ gsq,
    float* __restrict__ ebar)
{
  __shared__ __align__(16) _Float16 sW[11776];   // W1e | W2e | W3e (blob order)
  __shared__ __align__(16) _Float16 sBm[128 * 64];   // col-major; aliased as act2
  __shared__ __align__(16) _Float16 sAct1[128 * 96];
  __shared__ float sSeg[20], sSq[20];
  const _Float16* sW1 = sW;
  const _Float16* sW2 = sW + 5120;
  const _Float16* sW3 = sW + 9728;

  const int tid  = threadIdx.x;
  const int lane = tid & 31;
  const int wave = tid >> 5;
  const int r = blockIdx.x;
  const int b = blockIdx.y;
  const _Float16* xTb = xT + b * (NND * 32);

  // async-stage pre-converted weights (23.5KB) straight into LDS
  for (int i = tid; i < 1472; i += 256)
    async_g2l_b128(sW + i * 8, wblob + i * 8);
  // async-stage Bm columns: [0,32)=xT[r] (recv), [32,64)=xT[send[e]] (send)
  const int* sendr = send + r * EPN;
  for (int i = tid; i < 1024; i += 256) {
    const int col = i >> 3, c = i & 7;
    _Float16* dst = sBm + col * 64 + c * 8;
    if (col < EPN) {
      const int node = (c < 4) ? r : sendr[col];
      async_g2l_b128(dst, xTb + node * 32 + (c & 3) * 8);
    } else {
      unsigned* z = reinterpret_cast<unsigned*>(dst);
      z[0] = 0u; z[1] = 0u; z[2] = 0u; z[3] = 0u;
    }
  }
  for (int i = tid; i < 128 * 16; i += 256) {          // act1 K-pad rows 80..95
    const int col = i >> 4, k = 80 + (i & 15);
    sAct1[col * 96 + k] = (_Float16)0.f;
  }
  if (tid < 20) { sSeg[tid] = 0.f; sSq[tid] = 0.f; }
  wait_async0();
  __syncthreads();

  const int nb = wave * 16;
  const int h  = lane >> 4;
  const int cl = lane & 15;

#pragma unroll
  for (int mt = 0; mt < 5; ++mt) {                     // layer 1: 80 x 60
    v8f acc;
#pragma unroll
    for (int j = 0; j < 8; ++j) acc[j] = bfr1[mt * 16 + j + 8 * h];
#pragma unroll
    for (int kt = 0; kt < 2; ++kt) {
      v16h a  = frag_a(sW1, mt * 16, kt * 32, 64, lane);
      v16h bm = frag_b(sBm, nb,      kt * 32, 64, lane);
      acc = __builtin_amdgcn_wmma_f32_16x16x32_f16(false, a, false, bm,
                                                   (short)0, acc, false, false);
    }
#pragma unroll
    for (int j = 0; j < 8; ++j)
      sAct1[(nb + cl) * 96 + mt * 16 + j + 8 * h] = (_Float16)fmaxf(acc[j], 0.f);
  }
  __syncthreads();

  _Float16* sAct2 = sBm;                               // alias: Bm dead now
#pragma unroll
  for (int mt = 0; mt < 3; ++mt) {                     // layer 2: 40 x 80
    v8f acc;
#pragma unroll
    for (int j = 0; j < 8; ++j) {
      const int row = mt * 16 + j + 8 * h;
      acc[j] = (row < 40) ? bfr2[row] : 0.f;
    }
#pragma unroll
    for (int kt = 0; kt < 3; ++kt) {
      v16h a  = frag_a(sW2,   mt * 16, kt * 32, 96, lane);
      v16h bm = frag_b(sAct1, nb,      kt * 32, 96, lane);
      acc = __builtin_amdgcn_wmma_f32_16x16x32_f16(false, a, false, bm,
                                                   (short)0, acc, false, false);
    }
#pragma unroll
    for (int j = 0; j < 8; ++j)
      sAct2[(nb + cl) * 64 + mt * 16 + j + 8 * h] = (_Float16)fmaxf(acc[j], 0.f);
  }
  for (int i = tid; i < 128 * 16; i += 256) {          // act2 rows 48..63 = 0
    const int col = i >> 4, k = 48 + (i & 15);
    sAct2[col * 64 + k] = (_Float16)0.f;
  }
  __syncthreads();

#pragma unroll
  for (int mt = 0; mt < 2; ++mt) {                     // layer 3: 20 x 40
    v8f acc;
#pragma unroll
    for (int j = 0; j < 8; ++j) {
      const int row = mt * 16 + j + 8 * h;
      acc[j] = (row < 20) ? bfr3[row] : 0.f;
    }
#pragma unroll
    for (int kt = 0; kt < 2; ++kt) {
      v16h a  = frag_a(sW3,   mt * 16, kt * 32, 64, lane);
      v16h bm = frag_b(sAct2, nb,      kt * 32, 64, lane);
      acc = __builtin_amdgcn_wmma_f32_16x16x32_f16(false, a, false, bm,
                                                   (short)0, acc, false, false);
    }
    const bool colok = (nb + cl) < EPN;                // mask pad column 127
#pragma unroll
    for (int j = 0; j < 8; ++j) {
      const int row = mt * 16 + j + 8 * h;
      if (row < 20 && colok) {
        const float v = fmaxf(acc[j], 0.f);
        atomicAdd(&sSeg[row], v);
        atomicAdd(&sSq[row],  v * v);
      }
    }
  }
  __syncthreads();

  if (tid < 20) {
    ebar[(b * 20 + tid) * NND + r] = sSeg[tid];        // exclusive per (b,r)
    atomicAdd(&gsum[tid], sSeg[tid]);
    atomicAdd(&gsq[tid],  sSq[tid]);
  }
}

// ---- Kernel B: norm fold + node MLP 50->40->20->20 + head -----------------
__global__ __launch_bounds__(256) void node_mlp_kernel(
    const _Float16* __restrict__ xT, const _Float16* __restrict__ wblob_o,
    const float* __restrict__ bfo1, const float* __restrict__ bfo2,
    const float* __restrict__ bfo3,
    const float* __restrict__ Wfc,  const float* __restrict__ bfc,
    const float* __restrict__ gamma, const float* __restrict__ beta,
    const float* __restrict__ gsum, const float* __restrict__ gsq,
    const float* __restrict__ ebar, float* __restrict__ out)
{
  __shared__ __align__(16) _Float16 sWo[6144];   // W1o | W2o | W3o (blob order)
  __shared__ __align__(16) _Float16 sC[128 * 64];
  __shared__ __align__(16) _Float16 sA1[128 * 64];
  __shared__ __align__(16) _Float16 sA2[128 * 32];
  __shared__ float sSc[20], sSh[20], sNs[20];
  const _Float16* sW1 = sWo;
  const _Float16* sW2 = sWo + 3072;
  const _Float16* sW3 = sWo + 5120;

  const int tid  = threadIdx.x;
  const int lane = tid & 31;
  const int wave = tid >> 5;
  const int b = blockIdx.x;
  const _Float16* xTb = xT + b * (NND * 32);

  for (int i = tid; i < 768; i += 256)                 // async weight stage
    async_g2l_b128(sWo + i * 8, wblob_o + i * 8);
  for (int i = tid; i < 512; i += 256) {               // async C rows 0..31 = xT
    const int col = i >> 2, c = i & 3;
    async_g2l_b128(sC + col * 64 + c * 8, xTb + col * 32 + c * 8);
  }
  if (tid < 20) {
    const float Mtot = 64.f * 16256.f;                 // B * num_edges
    const float mu  = gsum[tid] / Mtot;
    const float var = gsq[tid] / Mtot - mu * mu;
    const float inv = rsqrtf(var + EPS_F);
    const float sc  = gamma[tid] * inv;
    sSc[tid] = sc;
    sSh[tid] = 127.f * (beta[tid] - sc * mu);
    sNs[tid] = 0.f;
  }
  for (int i = tid; i < 128 * 16; i += 256) {          // sA1 rows 48..63 = 0
    const int col = i >> 4, k = 48 + (i & 15);
    sA1[col * 64 + k] = (_Float16)0.f;
  }
  __syncthreads();                                     // sSc/sSh visible

  // C rows 32..63: normalized+folded Ebar (rows 52..63 zero)
  for (int i = tid; i < 128 * 32; i += 256) {
    const int col = i >> 5, k = 32 + (i & 31);
    float v = 0.f;
    if (k < 52) {
      const int c = k - 32;
      v = sSc[c] * ebar[(b * 20 + c) * NND + col] + sSh[c];
    }
    sC[col * 64 + k] = (_Float16)v;
  }
  wait_async0();
  __syncthreads();

  const int nb = wave * 16;
  const int h  = lane >> 4;
  const int cl = lane & 15;

#pragma unroll
  for (int mt = 0; mt < 3; ++mt) {                     // layer 1: 40 x 50
    v8f acc;
#pragma unroll
    for (int j = 0; j < 8; ++j) {
      const int row = mt * 16 + j + 8 * h;
      acc[j] = (row < 40) ? bfo1[row] : 0.f;
    }
#pragma unroll
    for (int kt = 0; kt < 2; ++kt) {
      v16h a  = frag_a(sW1, mt * 16, kt * 32, 64, lane);
      v16h bm = frag_b(sC,  nb,      kt * 32, 64, lane);
      acc = __builtin_amdgcn_wmma_f32_16x16x32_f16(false, a, false, bm,
                                                   (short)0, acc, false, false);
    }
#pragma unroll
    for (int j = 0; j < 8; ++j)
      sA1[(nb + cl) * 64 + mt * 16 + j + 8 * h] = (_Float16)fmaxf(acc[j], 0.f);
  }
  __syncthreads();

#pragma unroll
  for (int mt = 0; mt < 2; ++mt) {                     // layer 2: 20 x 40
    v8f acc;
#pragma unroll
    for (int j = 0; j < 8; ++j) {
      const int row = mt * 16 + j + 8 * h;
      acc[j] = (row < 20) ? bfo2[row] : 0.f;
    }
#pragma unroll
    for (int kt = 0; kt < 2; ++kt) {
      v16h a  = frag_a(sW2, mt * 16, kt * 32, 64, lane);
      v16h bm = frag_b(sA1, nb,      kt * 32, 64, lane);
      acc = __builtin_amdgcn_wmma_f32_16x16x32_f16(false, a, false, bm,
                                                   (short)0, acc, false, false);
    }
#pragma unroll
    for (int j = 0; j < 8; ++j)
      sA2[(nb + cl) * 32 + mt * 16 + j + 8 * h] = (_Float16)fmaxf(acc[j], 0.f);
  }
  __syncthreads();

#pragma unroll
  for (int mt = 0; mt < 2; ++mt) {                     // layer 3: 20 x 20 + sum
    v8f acc;
#pragma unroll
    for (int j = 0; j < 8; ++j) {
      const int row = mt * 16 + j + 8 * h;
      acc[j] = (row < 20) ? bfo3[row] : 0.f;
    }
    {
      v16h a  = frag_a(sW3, mt * 16, 0, 32, lane);
      v16h bm = frag_b(sA2, nb,      0, 32, lane);
      acc = __builtin_amdgcn_wmma_f32_16x16x32_f16(false, a, false, bm,
                                                   (short)0, acc, false, false);
    }
#pragma unroll
    for (int j = 0; j < 8; ++j) {
      const int row = mt * 16 + j + 8 * h;
      if (row < 20) atomicAdd(&sNs[row], fmaxf(acc[j], 0.f));
    }
  }
  __syncthreads();

  if (tid < 5) {                                       // head: Nsum @ Wfc.T + bfc
    float a = bfc[tid];
#pragma unroll
    for (int c = 0; c < 20; ++c) a += sNs[c] * Wfc[tid * 20 + c];
    out[b * 5 + tid] = a;
  }
}

extern "C" void kernel_launch(void* const* d_in, const int* in_sizes, int n_in,
                              void* d_out, int out_size, void* d_ws, size_t ws_size,
                              hipStream_t stream) {
  const float* x    = (const float*)d_in[0];
  const int*   send = (const int*)  d_in[2];
  const float* Wfr1 = (const float*)d_in[3];
  const float* bfr1 = (const float*)d_in[4];
  const float* Wfr2 = (const float*)d_in[5];
  const float* bfr2 = (const float*)d_in[6];
  const float* Wfr3 = (const float*)d_in[7];
  const float* bfr3 = (const float*)d_in[8];
  const float* gamma= (const float*)d_in[9];
  const float* beta = (const float*)d_in[10];
  const float* Wfo1 = (const float*)d_in[11];
  const float* bfo1 = (const float*)d_in[12];
  const float* Wfo2 = (const float*)d_in[13];
  const float* bfo2 = (const float*)d_in[14];
  const float* Wfo3 = (const float*)d_in[15];
  const float* bfo3 = (const float*)d_in[16];
  const float* Wfc  = (const float*)d_in[17];
  const float* bfc  = (const float*)d_in[18];

  float* ws   = (float*)d_ws;
  float* gsum = ws;
  float* gsq  = ws + 20;
  float* ebar = ws + 64;                       // 64*20*128 floats
  _Float16* hb = (_Float16*)(ws + WS_HALF_OFF);

  prep_kernel<<<NB, 256, 0, stream>>>(x, Wfr1, Wfr2, Wfr3, Wfo1, Wfo2, Wfo3, hb);
  zero_stats_kernel<<<1, 64, 0, stream>>>(ws);
  dim3 gA(NND, NB);
  edge_mlp_kernel<<<gA, 256, 0, stream>>>(hb + OFF_XT, send, hb + OFF_W1E,
                                          bfr1, bfr2, bfr3, gsum, gsq, ebar);
  node_mlp_kernel<<<NB, 256, 0, stream>>>(hb + OFF_XT, hb + OFF_W1O,
                                          bfo1, bfo2, bfo3, Wfc, bfc, gamma, beta,
                                          gsum, gsq, ebar, (float*)d_out);
}
